// GQA_5068061409417
// MI455X (gfx1250) — compile-verified, matrix operand order checked by reading, not run on hardware
//
#include <hip/hip_runtime.h>

typedef __bf16 v16bf __attribute__((ext_vector_type(16)));
typedef float  v8f   __attribute__((ext_vector_type(8)));
typedef unsigned u32x4 __attribute__((ext_vector_type(4)));
typedef int      i32x4 __attribute__((ext_vector_type(4)));
typedef int      i32x8 __attribute__((ext_vector_type(8)));

#define B_   2
#define S_   2048
#define D_   2048
#define QH_  32
#define KVH_ 8
#define HD_  64
// GROUPS = QH/KVH = 4 -> kv head = q head >> 2

static __device__ __forceinline__ __bf16 f2bf(float f) {
    // round-to-nearest-even fp32 -> bf16 via bit ops
    unsigned u = __builtin_bit_cast(unsigned, f);
    u += 0x7FFFu + ((u >> 16) & 1u);
    unsigned short h = (unsigned short)(u >> 16);
    return __builtin_bit_cast(__bf16, h);
}

static __device__ __forceinline__ v8f wmma_bf16(v16bf a, v16bf b, v8f c) {
    return __builtin_amdgcn_wmma_f32_16x16x32_bf16(false, a, false, b, (short)0, c,
                                                   false, false);
}

// ---------------------------------------------------------------------------
// TDM: DMA a 2D bf16 tile (tile_k x tile_rows, row stride in elements) from
// global memory into LDS. Builds a D# per CDNA5 ISA §8 (count=1, type=2,
// data_size=2B). Tracked with TENSORcnt.
// ---------------------------------------------------------------------------
static __device__ __forceinline__ void tdm_load_tile_2d(
    const __bf16* gsrc, unsigned lds_byte_off,
    unsigned tile_k, unsigned tile_rows, unsigned row_stride_elems)
{
    unsigned long long ga = (unsigned long long)(uintptr_t)gsrc;
    u32x4 g0;
    g0[0] = 1u;                                             // count=1, user D#
    g0[1] = lds_byte_off;                                   // lds_addr [63:32]
    g0[2] = (unsigned)(ga & 0xFFFFFFFFu);                   // global_addr lo
    g0[3] = (unsigned)((ga >> 32) & 0x1FFFFFFu)             // global_addr hi
          | 0x80000000u;                                    // type=2 [127:126]
    const unsigned dim0 = row_stride_elems;                 // tensor_dim0
    const unsigned dim1 = 1u << 20;                         // tensor_dim1 (ample)
    const unsigned long long st0 = row_stride_elems;        // tensor_dim0_stride
    i32x8 g1;
    g1[0] = (int)(1u << 16);                                // wg_mask=0, data_size=1 (2B)
    g1[1] = (int)((dim0 & 0xFFFFu) << 16);                  // dim0[15:0] @ bits 63:48
    g1[2] = (int)(((dim0 >> 16) & 0xFFFFu) | ((dim1 & 0xFFFFu) << 16));
    g1[3] = (int)(((dim1 >> 16) & 0xFFFFu) | ((tile_k & 0xFFFFu) << 16));   // tile_dim0
    g1[4] = (int)(tile_rows & 0xFFFFu);                     // tile_dim1, tile_dim2=0
    g1[5] = (int)(st0 & 0xFFFFFFFFu);                       // dim0_stride lo
    g1[6] = (int)((st0 >> 32) & 0xFFFFu);                   // dim0_stride hi
    g1[7] = 0;
    i32x4 z4 = {0, 0, 0, 0};
#if __clang_major__ >= 23
    i32x8 z8 = {0, 0, 0, 0, 0, 0, 0, 0};
    __builtin_amdgcn_tensor_load_to_lds(g0, g1, z4, z4, z8, 0);
#else
    __builtin_amdgcn_tensor_load_to_lds(g0, g1, z4, z4, 0);
#endif
}

static __device__ __forceinline__ unsigned lds_off_of(const void* p) {
    return (unsigned)(uintptr_t)p;   // flat LDS addr: low 32 bits = LDS offset
}

// ---------------------------------------------------------------------------
// Kernel 1: fp32 -> bf16 elementwise convert
// ---------------------------------------------------------------------------
__global__ void k_cvt(const float* __restrict__ in, __bf16* __restrict__ out, int n) {
    int i = blockIdx.x * blockDim.x + threadIdx.x;
    if (i < n) out[i] = f2bf(in[i]);
}

// ---------------------------------------------------------------------------
// Kernel 2: QKV projection (WMMA bf16, TDM-staged LDS tiles) + RMSNorm + RoPE
// grid: (S/64, QH + 2*KVH, B), block: 256 (8 waves)
// Each block: 64 s-rows x one 64-wide head; K staged in 64-wide chunks,
// double-buffered via TENSORcnt.
// ---------------------------------------------------------------------------
__global__ __launch_bounds__(256)
void k_qkv(const __bf16* __restrict__ xb,
           const __bf16* __restrict__ wqb,
           const __bf16* __restrict__ wkb,
           const __bf16* __restrict__ wvb,
           const float*  __restrict__ freq,
           const float*  __restrict__ qnw,
           const float*  __restrict__ knw,
           __bf16* __restrict__ qb,
           __bf16* __restrict__ kb,
           __bf16* __restrict__ vtb,
           float*  __restrict__ kout,
           float*  __restrict__ vout)
{
    __shared__ __align__(64) __bf16 sA[2][64][64];   // 16 KB
    __shared__ __align__(64) __bf16 sB[2][64][64];   // 16 KB
    __shared__ float tile[64][65];                   // 16.6 KB (epilogue)

    const int lane  = threadIdx.x & 31;
    const int wave  = threadIdx.x >> 5;
    const int bb    = blockIdx.z;
    const int hid   = blockIdx.y;
    const int m0    = blockIdx.x * 64;

    const __bf16* wb; int nbase;
    if (hid < QH_)             { wb = wqb; nbase = hid * HD_; }
    else if (hid < QH_ + KVH_) { wb = wkb; nbase = (hid - QH_) * HD_; }
    else                       { wb = wvb; nbase = (hid - QH_ - KVH_) * HD_; }

    const int tm    = wave >> 1;          // 0..3 (row tile)
    const int tn0   = (wave & 1) * 2;     // 0 or 2 (first col tile of pair)
    const int rhalf = lane >> 4;          // 0/1 -> K sub-chunk
    const int rl    = lane & 15;

    const __bf16* gA = xb + (size_t)(bb * S_ + m0) * D_;  // 64 rows x D
    const __bf16* gB = wb + (size_t)nbase * D_;           // 64 rows x D

    const int NC = D_ / 64;                               // 32 K-chunks
    if (wave == 0) {
        tdm_load_tile_2d(gA, lds_off_of(&sA[0][0][0]), 64, 64, D_);
        tdm_load_tile_2d(gB, lds_off_of(&sB[0][0][0]), 64, 64, D_);
    }

    v8f acc0 = {}; v8f acc1 = {};
    for (int c = 0; c < NC; ++c) {
        if (wave == 0) {
            if (c + 1 < NC) {
                const int nb = (c + 1) & 1;
                tdm_load_tile_2d(gA + (c + 1) * 64, lds_off_of(&sA[nb][0][0]), 64, 64, D_);
                tdm_load_tile_2d(gB + (c + 1) * 64, lds_off_of(&sB[nb][0][0]), 64, 64, D_);
                __builtin_amdgcn_s_wait_tensorcnt((short)2);   // chunk c complete
            } else {
                __builtin_amdgcn_s_wait_tensorcnt((short)0);
            }
        }
        __syncthreads();
        const int cb = c & 1;
#pragma unroll
        for (int kk = 0; kk < 64; kk += 32) {
            v16bf a  = *(const v16bf*)&sA[cb][tm * 16 + rl][kk + rhalf * 16];
            v16bf b0 = *(const v16bf*)&sB[cb][tn0 * 16 + rl][kk + rhalf * 16];
            v16bf b1 = *(const v16bf*)&sB[cb][(tn0 + 1) * 16 + rl][kk + rhalf * 16];
            acc0 = wmma_bf16(a, b0, acc0);
            acc1 = wmma_bf16(a, b1, acc1);
        }
        __syncthreads();   // done consuming before buffer is overwritten
    }

    // C layout: VGPR j of lane l -> row (j + 8*(l>=16)), col (l%16)
    const int r0 = tm * 16 + rhalf * 8;
#pragma unroll
    for (int j = 0; j < 8; ++j) {
        tile[r0 + j][tn0 * 16 + rl]       = acc0[j];
        tile[r0 + j][(tn0 + 1) * 16 + rl] = acc1[j];
    }
    __syncthreads();

    if (threadIdx.x < 64) {
        const int r  = threadIdx.x;
        const int sr = m0 + r;
        float f[64];
#pragma unroll
        for (int i = 0; i < 64; ++i) f[i] = tile[r][i];

        const bool isQ = hid < QH_;
        const bool isK = (hid >= QH_) && (hid < QH_ + KVH_);
        if (isQ || isK) {
            float ss = 0.f;
#pragma unroll
            for (int i = 0; i < 64; ++i) ss += f[i] * f[i];
            const float inv = rsqrtf(ss * (1.0f / 64.0f) + 1e-6f);
            const float* nw = isQ ? qnw : knw;
#pragma unroll
            for (int i = 0; i < 64; ++i) f[i] *= inv * nw[i];
            // interleaved-pair RoPE
#pragma unroll
            for (int i = 0; i < 32; ++i) {
                float ang = freq[sr * 32 + i];
                float co = __cosf(ang), si = __sinf(ang);
                float x1 = f[2 * i], x2 = f[2 * i + 1];
                f[2 * i]     = x1 * co - x2 * si;
                f[2 * i + 1] = x1 * si + x2 * co;
            }
        }

        if (isQ) {
            __bf16* dst = qb + (size_t)((bb * QH_ + hid) * S_ + sr) * HD_;
#pragma unroll
            for (int i = 0; i < 64; ++i) dst[i] = f2bf(f[i]);
        } else if (isK) {
            const int kvh = hid - QH_;
            const size_t base = (size_t)((bb * KVH_ + kvh) * S_ + sr) * HD_;
#pragma unroll
            for (int i = 0; i < 64; ++i) { kb[base + i] = f2bf(f[i]); kout[base + i] = f[i]; }
        } else {
            const int kvh = hid - QH_ - KVH_;
            const size_t base  = (size_t)((bb * KVH_ + kvh) * S_ + sr) * HD_;
            const size_t tbase = (size_t)(bb * KVH_ + kvh) * HD_ * S_;
#pragma unroll
            for (int i = 0; i < 64; ++i) {
                vout[base + i] = f[i];
                vtb[tbase + (size_t)i * S_ + sr] = f2bf(f[i]);   // transposed (hd, s)
            }
        }
    }
}

// ---------------------------------------------------------------------------
// Kernel 3: causal flash attention. grid: (S/128, QH, B), 8 waves/block,
// each wave owns a 16-row Q tile; 32-key tiles with online softmax.
// ---------------------------------------------------------------------------
__global__ __launch_bounds__(256)
void k_attn(const __bf16* __restrict__ qb,
            const __bf16* __restrict__ kb,
            const __bf16* __restrict__ vtb,
            __bf16* __restrict__ ob)
{
    __shared__ __align__(64) __bf16 pbuf[8][16][32];

    const int lane  = threadIdx.x & 31;
    const int wave  = threadIdx.x >> 5;
    const int bb    = blockIdx.z;
    const int qh    = blockIdx.y;
    const int kh    = qh >> 2;                    // GQA: 4 q-heads per kv-head
    const int qbase = blockIdx.x * 128 + wave * 16;
    const int rhalf = lane >> 4, rl = lane & 15;

    const __bf16* qrow = qb + (size_t)((bb * QH_ + qh) * S_ + qbase + rl) * HD_;
    const v16bf aq0 = *(const v16bf*)(qrow + rhalf * 16);        // hd 0..31
    const v16bf aq1 = *(const v16bf*)(qrow + 32 + rhalf * 16);   // hd 32..63

    const __bf16* kbase = kb  + (size_t)(bb * KVH_ + kh) * S_ * HD_;
    const __bf16* vbase = vtb + (size_t)(bb * KVH_ + kh) * HD_ * S_;

    float m_r[8], l_r[8];
    v8f accO[4] = {};
#pragma unroll
    for (int j = 0; j < 8; ++j) { m_r[j] = -INFINITY; l_r[j] = 0.f; }

    const int nkt = (qbase + 47) >> 5;            // key tiles covering 0..qbase+15
    for (int kt = 0; kt < nkt; ++kt) {
        const int k0 = kt * 32;
        // prefetch next K tile into near caches (WGP scope)
        __builtin_prefetch(kbase + (size_t)(k0 + 32 + rl) * HD_, 0, 3);

        // scores: S(16x32) = Q(16x64) @ K^T(64x32), two 16x16 sub-tiles
        const __bf16* kr0 = kbase + (size_t)(k0 + rl) * HD_ + rhalf * 16;
        const __bf16* kr1 = kbase + (size_t)(k0 + 16 + rl) * HD_ + rhalf * 16;
        v16bf bk00 = *(const v16bf*)(kr0);
        v16bf bk01 = *(const v16bf*)(kr0 + 32);
        v16bf bk10 = *(const v16bf*)(kr1);
        v16bf bk11 = *(const v16bf*)(kr1 + 32);
        v8f z = {};
        v8f sc0 = wmma_bf16(aq1, bk01, wmma_bf16(aq0, bk00, z));
        v8f sc1 = wmma_bf16(aq1, bk11, wmma_bf16(aq0, bk10, z));

        // online softmax (rows span 16 lanes within each half-wave)
#pragma unroll
        for (int j = 0; j < 8; ++j) {
            const int qr = qbase + j + rhalf * 8;
            float s0 = sc0[j] * 0.125f;           // HD^-0.5
            float s1 = sc1[j] * 0.125f;
            if (k0 + rl > qr)      s0 = -INFINITY;
            if (k0 + 16 + rl > qr) s1 = -INFINITY;
            float t = fmaxf(s0, s1);
            t = fmaxf(t, __shfl_xor(t, 1, 32));
            t = fmaxf(t, __shfl_xor(t, 2, 32));
            t = fmaxf(t, __shfl_xor(t, 4, 32));
            t = fmaxf(t, __shfl_xor(t, 8, 32));
            const float nm   = fmaxf(m_r[j], t);
            const float corr = __expf(m_r[j] - nm);
            const float p0 = __expf(s0 - nm);
            const float p1 = __expf(s1 - nm);
            float ps = p0 + p1;
            ps += __shfl_xor(ps, 1, 32);
            ps += __shfl_xor(ps, 2, 32);
            ps += __shfl_xor(ps, 4, 32);
            ps += __shfl_xor(ps, 8, 32);
            l_r[j] = l_r[j] * corr + ps;
            m_r[j] = nm;
#pragma unroll
            for (int c = 0; c < 4; ++c) accO[c][j] *= corr;
            pbuf[wave][j + rhalf * 8][rl]      = f2bf(p0);
            pbuf[wave][j + rhalf * 8][16 + rl] = f2bf(p1);
        }
        asm volatile("s_wait_dscnt 0x0" ::: "memory");

        // A-fragment of P (16x32) from per-wave LDS slab
        const v16bf ap = *(const v16bf*)&pbuf[wave][rl][rhalf * 16];

        // accO += P(16x32) @ V(32x64), V read transposed (hd-major)
#pragma unroll
        for (int c = 0; c < 4; ++c) {
            const __bf16* vr = vbase + (size_t)(c * 16 + rl) * S_ + k0 + rhalf * 16;
            v16bf bv = *(const v16bf*)vr;
            accO[c] = wmma_bf16(ap, bv, accO[c]);
        }
    }

    // finalize: divide by l, write bf16 attn-out (b, s, QH*HD)
#pragma unroll
    for (int j = 0; j < 8; ++j) {
        const float inv = 1.0f / l_r[j];
        const int row = qbase + j + rhalf * 8;
        __bf16* dst = ob + (size_t)(bb * S_ + row) * (QH_ * HD_) + qh * HD_;
#pragma unroll
        for (int c = 0; c < 4; ++c) dst[c * 16 + rl] = f2bf(accO[c][j] * inv);
    }
}

// ---------------------------------------------------------------------------
// Kernel 4: output projection out = attn_out @ wo^T (fp32), TDM-staged tiles
// grid: (B*S/64, D/64), 8 waves x two 16x16 tiles
// ---------------------------------------------------------------------------
__global__ __launch_bounds__(256)
void k_oproj(const __bf16* __restrict__ ob,
             const __bf16* __restrict__ wob,
             float* __restrict__ out)
{
    __shared__ __align__(64) __bf16 sA[2][64][64];
    __shared__ __align__(64) __bf16 sB[2][64][64];

    const int lane  = threadIdx.x & 31;
    const int wave  = threadIdx.x >> 5;
    const int tm    = wave >> 1;
    const int tn0   = (wave & 1) * 2;
    const int rhalf = lane >> 4, rl = lane & 15;
    const int m0    = blockIdx.x * 64;
    const int n0    = blockIdx.y * 64;

    const __bf16* gA = ob  + (size_t)m0 * D_;
    const __bf16* gB = wob + (size_t)n0 * D_;

    const int NC = D_ / 64;
    if (wave == 0) {
        tdm_load_tile_2d(gA, lds_off_of(&sA[0][0][0]), 64, 64, D_);
        tdm_load_tile_2d(gB, lds_off_of(&sB[0][0][0]), 64, 64, D_);
    }

    v8f acc0 = {}, acc1 = {};
    for (int c = 0; c < NC; ++c) {
        if (wave == 0) {
            if (c + 1 < NC) {
                const int nb = (c + 1) & 1;
                tdm_load_tile_2d(gA + (c + 1) * 64, lds_off_of(&sA[nb][0][0]), 64, 64, D_);
                tdm_load_tile_2d(gB + (c + 1) * 64, lds_off_of(&sB[nb][0][0]), 64, 64, D_);
                __builtin_amdgcn_s_wait_tensorcnt((short)2);
            } else {
                __builtin_amdgcn_s_wait_tensorcnt((short)0);
            }
        }
        __syncthreads();
        const int cb = c & 1;
#pragma unroll
        for (int kk = 0; kk < 64; kk += 32) {
            v16bf a  = *(const v16bf*)&sA[cb][tm * 16 + rl][kk + rhalf * 16];
            v16bf b0 = *(const v16bf*)&sB[cb][tn0 * 16 + rl][kk + rhalf * 16];
            v16bf b1 = *(const v16bf*)&sB[cb][(tn0 + 1) * 16 + rl][kk + rhalf * 16];
            acc0 = wmma_bf16(a, b0, acc0);
            acc1 = wmma_bf16(a, b1, acc1);
        }
        __syncthreads();
    }

    const int r = m0 + tm * 16 + rhalf * 8;
#pragma unroll
    for (int j = 0; j < 8; ++j) {
        out[(size_t)(r + j) * D_ + n0 + tn0 * 16 + rl]       = acc0[j];
        out[(size_t)(r + j) * D_ + n0 + (tn0 + 1) * 16 + rl] = acc1[j];
    }
}

// ---------------------------------------------------------------------------
extern "C" void kernel_launch(void* const* d_in, const int* in_sizes, int n_in,
                              void* d_out, int out_size, void* d_ws, size_t ws_size,
                              hipStream_t stream) {
    (void)in_sizes; (void)n_in; (void)out_size; (void)ws_size;
    const float* x    = (const float*)d_in[0];
    const float* freq = (const float*)d_in[1];
    const float* wq   = (const float*)d_in[2];
    const float* wk   = (const float*)d_in[3];
    const float* wv   = (const float*)d_in[4];
    const float* wo   = (const float*)d_in[5];
    const float* qnw  = (const float*)d_in[6];
    const float* knw  = (const float*)d_in[7];
    float* out = (float*)d_out;

    __bf16* ws = (__bf16*)d_ws;
    size_t off = 0;
    __bf16* xb  = ws + off; off += (size_t)B_ * S_ * D_;          // x bf16
    __bf16* wqb = ws + off; off += (size_t)QH_ * HD_ * D_;        // wq bf16
    __bf16* wkb = ws + off; off += (size_t)KVH_ * HD_ * D_;       // wk bf16
    __bf16* wvb = ws + off; off += (size_t)KVH_ * HD_ * D_;       // wv bf16
    __bf16* wob = ws + off; off += (size_t)D_ * QH_ * HD_;        // wo bf16
    __bf16* qb  = ws + off; off += (size_t)B_ * QH_ * S_ * HD_;   // q (normed+rope)
    __bf16* kb  = ws + off; off += (size_t)B_ * KVH_ * S_ * HD_;  // k (normed+rope)
    __bf16* vtb = ws + off; off += (size_t)B_ * KVH_ * HD_ * S_;  // v transposed
    __bf16* ob  = ws + off; off += (size_t)B_ * S_ * QH_ * HD_;   // attn out

    float* kout = out  + (size_t)B_ * S_ * D_;             // k_new fp32
    float* vout = kout + (size_t)B_ * KVH_ * S_ * HD_;     // v_new fp32

    const size_t nx  = (size_t)B_ * S_ * D_;
    const size_t nwq = (size_t)QH_ * HD_ * D_;
    const size_t nwk = (size_t)KVH_ * HD_ * D_;
    const size_t nwo = (size_t)D_ * QH_ * HD_;

    k_cvt<<<dim3((unsigned)((nx  + 255) / 256)), dim3(256), 0, stream>>>(x,  xb,  (int)nx);
    k_cvt<<<dim3((unsigned)((nwq + 255) / 256)), dim3(256), 0, stream>>>(wq, wqb, (int)nwq);
    k_cvt<<<dim3((unsigned)((nwk + 255) / 256)), dim3(256), 0, stream>>>(wk, wkb, (int)nwk);
    k_cvt<<<dim3((unsigned)((nwk + 255) / 256)), dim3(256), 0, stream>>>(wv, wvb, (int)nwk);
    k_cvt<<<dim3((unsigned)((nwo + 255) / 256)), dim3(256), 0, stream>>>(wo, wob, (int)nwo);

    k_qkv<<<dim3(S_ / 64, QH_ + 2 * KVH_, B_), dim3(256), 0, stream>>>(
        xb, wqb, wkb, wvb, freq, qnw, knw, qb, kb, vtb, kout, vout);

    k_attn<<<dim3(S_ / 128, QH_, B_), dim3(256), 0, stream>>>(qb, kb, vtb, ob);

    k_oproj<<<dim3((B_ * S_) / 64, D_ / 64, 1), dim3(256), 0, stream>>>(ob, wob, out);
}
